// SingleHeadTansformer_18708877541425
// MI455X (gfx1250) — compile-verified, matrix operand order checked by reading, not run on hardware
//
#include <hip/hip_runtime.h>
#include <hip/hip_bf16.h>

// ---------------------------------------------------------------------------
// Problem constants (from reference): B=8, S=2048, IN=256, D=1024, OUT=256
// Only attended[:, -1, :] is consumed -> only last-row Q / last-row attention
// is needed. Heavy work is 3 GEMMs: h = x@Wi, K = h@Wk, V = h@Wv.
// ---------------------------------------------------------------------------
#define BATCH   8
#define SEQ     2048
#define INDIM   256
#define DDIM    1024
#define OUTDIM  256
#define MROWS   (BATCH * SEQ)   // 16384

typedef __bf16 bf16_t;
typedef __attribute__((ext_vector_type(16))) __bf16 v16bf;
typedef __attribute__((ext_vector_type(8)))  __bf16 v8bf;
typedef __attribute__((ext_vector_type(8)))  float  v8f;
typedef __attribute__((ext_vector_type(4)))  unsigned int v4u;
typedef __attribute__((ext_vector_type(8)))  int v8i;
typedef __attribute__((ext_vector_type(4)))  int v4i;

static __device__ __forceinline__ bf16_t f2bf(float f) {
    union { unsigned short u; bf16_t b; } c;
    unsigned x = __float_as_uint(f);
    x += 0x7FFFu + ((x >> 16) & 1u);      // round-to-nearest-even
    c.u = (unsigned short)(x >> 16);
    return c.b;
}
static __device__ __forceinline__ float bf2f(bf16_t b) {
    union { bf16_t b; unsigned short u; } c; c.b = b;
    return __uint_as_float(((unsigned)c.u) << 16);
}
// Generic pointer to LDS: low 32 bits are the workgroup-relative byte offset.
static __device__ __forceinline__ unsigned lds_off32(const void* p) {
    return (unsigned)(uintptr_t)p;
}

// ---------------------------------------------------------------------------
// Conversion kernels
// ---------------------------------------------------------------------------
__global__ void cvt_f32_to_bf16(const float* __restrict__ src,
                                bf16_t* __restrict__ dst, int n) {
    int i = blockIdx.x * blockDim.x + threadIdx.x;
    if (i < n) dst[i] = f2bf(src[i]);
}

// W is (Kd x Nd) row-major fp32; write Wt as (Nd x Kd) row-major bf16.
__global__ void transpose_cvt_bf16(const float* __restrict__ W,
                                   bf16_t* __restrict__ Wt, int Kd, int Nd) {
    int i = blockIdx.x * blockDim.x + threadIdx.x;
    if (i < Kd * Nd) {
        int k = i / Nd, n = i % Nd;                 // coalesced read
        Wt[(size_t)n * Kd + k] = f2bf(W[i]);
    }
}

// ---------------------------------------------------------------------------
// TDM helper: async-copy a 2D tile (rows x cols bf16 elements, global row
// stride = stride_elems) into LDS at byte offset lds_off. The TDM pad feature
// inserts 32B after every 64B so the LDS row pitch is 96B (= 48 elements),
// matching the padded LDS tile layout used by the WMMA fragment loads.
// Tracked by this wave's TENSORcnt.
// This toolchain exposes the 6-arg builtin:
//   (uint32x4 g0, int32x8 g1, int32x4 g2, int32x4 g3, int32x8 gx, i32 cpol)
// ---------------------------------------------------------------------------
#if __has_builtin(__builtin_amdgcn_tensor_load_to_lds)
#define HAVE_TDM 1
static __device__ __forceinline__ void tdm_load_tile(
        unsigned lds_off, const bf16_t* gptr,
        unsigned rows, unsigned cols, unsigned stride_elems) {
    unsigned long long ga = (unsigned long long)(uintptr_t)gptr;
    v4u g0;
    g0.x = 0x1u;                                      // count=1 valid user D#
    g0.y = lds_off;                                   // LDS byte address
    g0.z = (unsigned)(ga & 0xFFFFFFFFu);              // global_addr[31:0]
    g0.w = (unsigned)((ga >> 32) & 0x1FFFFFFu)        // global_addr[56:32]
         | (2u << 30);                                // type = 2 ("image")
    v8i g1;
    g1[0] = (int)((1u << 16)                          // data_size = 2 bytes
                | (1u << 20)                          // pad_enable
                | (3u << 22)                          // pad_interval: 64B
                | (7u << 25));                        // pad_amount: 32B
    g1[1] = (int)((cols & 0xFFFFu) << 16);            // tensor_dim0[15:0]
    g1[2] = (int)(((cols >> 16) & 0xFFFFu)            // tensor_dim0[31:16]
                | ((rows & 0xFFFFu) << 16));          // tensor_dim1[15:0]
    g1[3] = (int)(((rows >> 16) & 0xFFFFu)            // tensor_dim1[31:16]
                | ((cols & 0xFFFFu) << 16));          // tile_dim0
    g1[4] = (int)(rows & 0xFFFFu);                    // tile_dim1 (tile_dim2=0)
    g1[5] = (int)stride_elems;                        // tensor_dim0_stride[31:0]
    g1[6] = 0;                                        // stride hi / dim1_stride
    g1[7] = 0;
    const v4i gz4 = {0, 0, 0, 0};                     // groups 2/3: 2D tensor
    const v8i gz8 = {0, 0, 0, 0, 0, 0, 0, 0};
    __builtin_amdgcn_tensor_load_to_lds(g0, g1, gz4, gz4, gz8, 0);
}
#else
#define HAVE_TDM 0
#endif

// ---------------------------------------------------------------------------
// bf16 WMMA GEMM: C(MxN, bf16) = A(MxK, bf16) * Bt(NxK, bf16)^T + bias(N, f32)
// Block tile 128x128, BK=32. 256 threads = 8 waves in a 2(M) x 4(N) grid;
// each wave computes 64x32 = 4x2 tiles of v_wmma_f32_16x16x32_bf16.
// Tiles are staged by the Tensor Data Mover (per-wave 16x32 slabs, double-
// buffered LDS, TENSORcnt-synchronized), so the compute loop issues only
// ds_load_b128 + v_wmma.
// M % 128 == 0, N % 128 == 0, K % 32 == 0 (true for all call sites).
// ---------------------------------------------------------------------------
#define BM 128
#define BN 128
#define BK 32
#define LDSP 48   // LDS row pitch in elements: 96B (16B-aligned, spreads banks)

__global__ __launch_bounds__(256)
void gemm_bf16_wmma(const bf16_t* __restrict__ A,
                    const bf16_t* __restrict__ Bt,
                    const float*  __restrict__ bias,
                    bf16_t* __restrict__ C,
                    int M, int N, int K) {
    __shared__ bf16_t As[2][BM][LDSP];
    __shared__ bf16_t Bs[2][BN][LDSP];

    const int tid  = threadIdx.x;
    const int lane = tid & 31;
    const int wave = tid >> 5;          // 0..7
    const int wm   = wave >> 2;         // 0..1  -> wave M offset = wm*64
    const int wn   = wave & 3;          // 0..3  -> wave N offset = wn*32

    const int block_m = blockIdx.y * BM;
    const int block_n = blockIdx.x * BN;

    const int half = lane >> 4;         // 0 or 1
    const int l16  = lane & 15;
    // 16-bit A 16x32 layout: lanes 0-15 hold K0..7 (v0..3) and K16..23 (v4..7)
    //                        lanes 16-31 hold K8..15 and K24..31
    const int k0 = half * 8;
    const int k1 = 16 + half * 8;

    v8f acc[4][2];
    const v8f vzero = {0.f,0.f,0.f,0.f,0.f,0.f,0.f,0.f};
    for (int i = 0; i < 4; ++i)
        for (int j = 0; j < 2; ++j)
            acc[i][j] = vzero;

    const int nk = K / BK;

#if HAVE_TDM
    // Per-wave slab: 16 rows of A (and of B) per K-step, issued via TDM.
    const int srow = wave * 16;
    const bf16_t* Ag = A  + (size_t)(block_m + srow) * K;
    const bf16_t* Bg = Bt + (size_t)(block_n + srow) * K;

    // Prologue: fill both buffers.
    tdm_load_tile(lds_off32(&As[0][srow][0]), Ag, 16, BK, (unsigned)K);
    tdm_load_tile(lds_off32(&Bs[0][srow][0]), Bg, 16, BK, (unsigned)K);
    if (nk > 1) {
        tdm_load_tile(lds_off32(&As[1][srow][0]), Ag + BK, 16, BK, (unsigned)K);
        tdm_load_tile(lds_off32(&Bs[1][srow][0]), Bg + BK, 16, BK, (unsigned)K);
    }
#endif

    for (int kt = 0; kt < nk; ++kt) {
        const int p = kt & 1;
#if HAVE_TDM
        // Wait for *this wave's* slabs of step kt (in-order completion: the
        // two ops for kt+1, if issued, may still be outstanding -> wait <= 2).
        if (kt + 1 < nk) __builtin_amdgcn_s_wait_tensorcnt(2);
        else             __builtin_amdgcn_s_wait_tensorcnt(0);
        __syncthreads();   // now every wave's slabs for kt are visible
#else
        // Fallback: manual staging, 2 chunks of 8 bf16 per thread per tile.
#pragma unroll
        for (int c = 0; c < 2; ++c) {
            int chunk = tid * 2 + c;          // 0..511
            int m  = chunk >> 2;              // BK/8 = 4 chunks per row
            int kk = (chunk & 3) * 8;
            *(v8bf*)(&As[p][m][kk]) =
                *(const v8bf*)(A  + (size_t)(block_m + m) * K + kt * BK + kk);
            *(v8bf*)(&Bs[p][m][kk]) =
                *(const v8bf*)(Bt + (size_t)(block_n + m) * K + kt * BK + kk);
        }
        __syncthreads();
#endif

        // B fragments for this wave's two 16-wide column tiles
        v16bf bfrag[2];
#pragma unroll
        for (int j = 0; j < 2; ++j) {
            int n = wn * 32 + j * 16 + l16;
            v8bf lo = *(const v8bf*)(&Bs[p][n][k0]);
            v8bf hi = *(const v8bf*)(&Bs[p][n][k1]);
            bfrag[j] = __builtin_shufflevector(lo, hi,
                        0,1,2,3,4,5,6,7,8,9,10,11,12,13,14,15);
        }
#pragma unroll
        for (int i = 0; i < 4; ++i) {
            int m = wm * 64 + i * 16 + l16;
            v8bf lo = *(const v8bf*)(&As[p][m][k0]);
            v8bf hi = *(const v8bf*)(&As[p][m][k1]);
            v16bf afrag = __builtin_shufflevector(lo, hi,
                        0,1,2,3,4,5,6,7,8,9,10,11,12,13,14,15);
#pragma unroll
            for (int j = 0; j < 2; ++j) {
                acc[i][j] = __builtin_amdgcn_wmma_f32_16x16x32_bf16(
                    false, afrag, false, bfrag[j],
                    (short)0, acc[i][j], false, false);
            }
        }
        __syncthreads();   // all waves done reading buffer p

#if HAVE_TDM
        if (kt + 2 < nk) { // refill buffer p for step kt+2
            tdm_load_tile(lds_off32(&As[p][srow][0]), Ag + (kt + 2) * BK,
                          16, BK, (unsigned)K);
            tdm_load_tile(lds_off32(&Bs[p][srow][0]), Bg + (kt + 2) * BK,
                          16, BK, (unsigned)K);
        }
#endif
    }

    // Epilogue: C/D layout: VGPR r -> M = r (lanes 0-15) or 8+r (lanes 16-31),
    // N = lane&15. Fused bias add + bf16 store.
#pragma unroll
    for (int i = 0; i < 4; ++i) {
        int mbase = block_m + wm * 64 + i * 16 + half * 8;
#pragma unroll
        for (int j = 0; j < 2; ++j) {
            int n = block_n + wn * 32 + j * 16 + l16;
            float bv = bias ? bias[n] : 0.0f;
#pragma unroll
            for (int r = 0; r < 8; ++r) {
                C[(size_t)(mbase + r) * N + n] = f2bf(acc[i][j][r] + bv);
            }
        }
    }
}

// ---------------------------------------------------------------------------
// q_last[b][d] = h[b, S-1, :] . Wq[:, d] + bq[d]      (8 x 1024, trivial work)
// ---------------------------------------------------------------------------
__global__ __launch_bounds__(256)
void qlast_kernel(const bf16_t* __restrict__ h, const float* __restrict__ Wq,
                  const float* __restrict__ bq, float* __restrict__ q) {
    int i = blockIdx.x * blockDim.x + threadIdx.x;    // 0..8191
    int b = i >> 10, d = i & (DDIM - 1);
    const bf16_t* hrow = h + ((size_t)b * SEQ + (SEQ - 1)) * DDIM;
    float acc = bq[d];
    for (int k = 0; k < DDIM; ++k)
        acc += bf2f(hrow[k]) * Wq[(size_t)k * DDIM + d];   // coalesced over d
    q[i] = acc;
}

// ---------------------------------------------------------------------------
// Last-row attention per batch: scores -> softmax -> attended (one WG / batch)
// ---------------------------------------------------------------------------
__global__ __launch_bounds__(256)
void attn_last_kernel(const float* __restrict__ q, const bf16_t* __restrict__ Km,
                      const bf16_t* __restrict__ Vm, float* __restrict__ att) {
    const int b = blockIdx.x, tid = threadIdx.x;
    __shared__ float qs[DDIM];
    __shared__ float sc[SEQ];
    __shared__ float red[256];

    for (int d = tid; d < DDIM; d += 256) qs[d] = q[b * DDIM + d];
    __syncthreads();

    const float scale = 0.03125f;   // 1/sqrt(1024)
    for (int j = tid; j < SEQ; j += 256) {
        const bf16_t* krow = Km + ((size_t)b * SEQ + j) * DDIM;
        float a = 0.f;
        for (int k = 0; k < DDIM; ++k) a += qs[k] * bf2f(krow[k]);
        sc[j] = a * scale;
    }
    __syncthreads();

    float mx = -3.0e38f;
    for (int j = tid; j < SEQ; j += 256) mx = fmaxf(mx, sc[j]);
    red[tid] = mx; __syncthreads();
    for (int off = 128; off > 0; off >>= 1) {
        if (tid < off) red[tid] = fmaxf(red[tid], red[tid + off]);
        __syncthreads();
    }
    const float m = red[0];
    __syncthreads();

    float sm = 0.f;
    for (int j = tid; j < SEQ; j += 256) {
        float e = __expf(sc[j] - m);
        sc[j] = e; sm += e;
    }
    red[tid] = sm; __syncthreads();
    for (int off = 128; off > 0; off >>= 1) {
        if (tid < off) red[tid] += red[tid + off];
        __syncthreads();
    }
    const float inv = 1.0f / red[0];
    __syncthreads();

    float a0 = 0.f, a1 = 0.f, a2 = 0.f, a3 = 0.f;
    for (int j = 0; j < SEQ; ++j) {
        float w = sc[j] * inv;                     // LDS broadcast
        const bf16_t* vrow = Vm + ((size_t)b * SEQ + j) * DDIM;
        a0 += w * bf2f(vrow[tid]);
        a1 += w * bf2f(vrow[tid + 256]);
        a2 += w * bf2f(vrow[tid + 512]);
        a3 += w * bf2f(vrow[tid + 768]);
    }
    float* arow = att + (size_t)b * DDIM;
    arow[tid] = a0; arow[tid + 256] = a1; arow[tid + 512] = a2; arow[tid + 768] = a3;
}

// ---------------------------------------------------------------------------
// out[b][o] = att[b,:] . Wo[:,o] + bo[o]            (8 x 256, trivial)
// ---------------------------------------------------------------------------
__global__ __launch_bounds__(256)
void out_kernel(const float* __restrict__ att, const float* __restrict__ Wo,
                const float* __restrict__ bo, float* __restrict__ out) {
    int i = blockIdx.x * blockDim.x + threadIdx.x;   // 0..2047
    int b = i >> 8, o = i & (OUTDIM - 1);
    const float* arow = att + (size_t)b * DDIM;
    float acc = bo[o];
    for (int k = 0; k < DDIM; ++k)
        acc += arow[k] * Wo[(size_t)k * OUTDIM + o];
    out[i] = acc;
}

// ---------------------------------------------------------------------------
extern "C" void kernel_launch(void* const* d_in, const int* in_sizes, int n_in,
                              void* d_out, int out_size, void* d_ws, size_t ws_size,
                              hipStream_t stream) {
    (void)in_sizes; (void)n_in; (void)out_size; (void)ws_size;

    const float* x  = (const float*)d_in[0];
    const float* Wi = (const float*)d_in[1];
    const float* bi = (const float*)d_in[2];
    const float* Wq = (const float*)d_in[3];
    const float* bq = (const float*)d_in[4];
    const float* Wk = (const float*)d_in[5];
    const float* bk = (const float*)d_in[6];
    const float* Wv = (const float*)d_in[7];
    const float* bv = (const float*)d_in[8];
    const float* Wo = (const float*)d_in[9];
    const float* bo = (const float*)d_in[10];
    float* out = (float*)d_out;

    // Workspace layout (bf16 region first, then f32 region) — ~114 MB.
    bf16_t* xb  = (bf16_t*)d_ws;                    // MROWS x INDIM
    bf16_t* WiT = xb  + (size_t)MROWS * INDIM;      // DDIM x INDIM (transposed)
    bf16_t* WkT = WiT + (size_t)INDIM * DDIM;       // DDIM x DDIM (transposed)
    bf16_t* WvT = WkT + (size_t)DDIM * DDIM;        // DDIM x DDIM (transposed)
    bf16_t* h   = WvT + (size_t)DDIM * DDIM;        // MROWS x DDIM
    bf16_t* Km  = h   + (size_t)MROWS * DDIM;       // MROWS x DDIM
    bf16_t* Vm  = Km  + (size_t)MROWS * DDIM;       // MROWS x DDIM
    float*  qv  = (float*)(Vm + (size_t)MROWS * DDIM);  // BATCH x DDIM
    float*  att = qv + (size_t)BATCH * DDIM;            // BATCH x DDIM

    // 1) precision conversion / weight transposition
    {
        int n = MROWS * INDIM;
        cvt_f32_to_bf16<<<(n + 255) / 256, 256, 0, stream>>>(x, xb, n);
        transpose_cvt_bf16<<<(INDIM * DDIM + 255) / 256, 256, 0, stream>>>(Wi, WiT, INDIM, DDIM);
        transpose_cvt_bf16<<<(DDIM * DDIM + 255) / 256, 256, 0, stream>>>(Wk, WkT, DDIM, DDIM);
        transpose_cvt_bf16<<<(DDIM * DDIM + 255) / 256, 256, 0, stream>>>(Wv, WvT, DDIM, DDIM);
    }

    // 2) heavy GEMMs on the WMMA + TDM path
    dim3 blk(256);
    dim3 grid_h(DDIM / BN, MROWS / BM);   // (8, 128)
    gemm_bf16_wmma<<<grid_h, blk, 0, stream>>>(xb, WiT, bi, h,  MROWS, DDIM, INDIM);
    gemm_bf16_wmma<<<grid_h, blk, 0, stream>>>(h,  WkT, bk, Km, MROWS, DDIM, DDIM);
    gemm_bf16_wmma<<<grid_h, blk, 0, stream>>>(h,  WvT, bv, Vm, MROWS, DDIM, DDIM);

    // 3) last-row Q, attention, output projection (all tiny)
    qlast_kernel<<<(BATCH * DDIM) / 256, 256, 0, stream>>>(h, Wq, bq, qv);
    attn_last_kernel<<<BATCH, 256, 0, stream>>>(qv, Km, Vm, att);
    out_kernel<<<(BATCH * OUTDIM) / 256, 256, 0, stream>>>(att, Wo, bo, out);
}
